// SNPReductionNetModel_80144089743473
// MI455X (gfx1250) — matrix operand order repaired and checked by reading, not compiled, not run on hardware
//
#include <hip/hip_runtime.h>
#include <hip/hip_bf16.h>

// ---------------------------------------------------------------------------
// Problem constants (from the reference)
// ---------------------------------------------------------------------------
#define BATCH   512
#define NBLK    4096          // N_BLOCKS
#define KSP     64            // BLOCK*BITS, sparse dot length
#define IN_DIM  (NBLK * KSP)  // 262144
#define D0      1024
#define D1      512
#define LN_EPS  1e-5f

typedef __attribute__((ext_vector_type(16))) __bf16 v16bf;
typedef __attribute__((ext_vector_type(8)))  __bf16 v8bf;
typedef __attribute__((ext_vector_type(8)))  float  v8f;

union AFrag { v16bf v; v8bf h[2]; };

// ---------------------------------------------------------------------------
// 0) Transpose f32 (K x N, row-major) -> bf16 (N x K, row-major).
//    Puts weights in "Bt" layout so the WMMA B fragment is contiguous loads.
// ---------------------------------------------------------------------------
__global__ void transpose_to_bf16_kernel(const float* __restrict__ src,
                                         __bf16* __restrict__ dst,
                                         int K, int N) {
    int id = blockIdx.x * 256 + threadIdx.x;   // id = n*K + k (dst-linear)
    if (id >= K * N) return;
    int n = id / K;
    int k = id - n * K;
    dst[id] = (__bf16)src[(size_t)k * N + n];
}

// ---------------------------------------------------------------------------
// 1) Gather-dot: g[b,n] = sum_k x[b, idx[n,k]] * w_sparse[n,k]
//    idx groups of 4 are contiguous (snp*4 + 0..3) -> aligned float4 gathers.
// ---------------------------------------------------------------------------
__global__ void gather_dot_kernel(const float* __restrict__ x,
                                  const int*   __restrict__ idx,
                                  const float* __restrict__ wsp,
                                  float*       __restrict__ g) {
    int t = blockIdx.x * 256 + threadIdx.x;    // 0 .. BATCH*NBLK-1
    int b = t >> 12;                           // / NBLK
    int n = t & (NBLK - 1);
    const float* xrow = x   + (size_t)b * IN_DIM;
    const int*   irow = idx + n * KSP;
    const float* wrow = wsp + n * KSP;
    float acc = 0.0f;
#pragma unroll
    for (int j = 0; j < KSP / 4; ++j) {
        int base  = irow[4 * j];               // multiple of 4 -> 16B aligned
        float4 xv = *(const float4*)(xrow + base);
        float4 wv = *(const float4*)(wrow + 4 * j);
        acc += xv.x * wv.x + xv.y * wv.y + xv.z * wv.z + xv.w * wv.w;
    }
    g[t] = acc;
}

// ---------------------------------------------------------------------------
// 2) LayerNorm over the NBLK axis per batch row; emit bf16 activations.
// ---------------------------------------------------------------------------
__global__ void layernorm_kernel(const float* __restrict__ g,
                                 const float* __restrict__ ln_g,
                                 const float* __restrict__ ln_b,
                                 __bf16*      __restrict__ h) {
    __shared__ float s_sum[256];
    __shared__ float s_sq[256];
    int b = blockIdx.x;
    const float* grow = g + (size_t)b * NBLK;
    float sum = 0.0f, sq = 0.0f;
    for (int n = threadIdx.x; n < NBLK; n += 256) {
        float v = grow[n];
        sum += v;
        sq  += v * v;
    }
    s_sum[threadIdx.x] = sum;
    s_sq[threadIdx.x]  = sq;
    __syncthreads();
    for (int st = 128; st > 0; st >>= 1) {
        if (threadIdx.x < st) {
            s_sum[threadIdx.x] += s_sum[threadIdx.x + st];
            s_sq[threadIdx.x]  += s_sq[threadIdx.x + st];
        }
        __syncthreads();
    }
    float mu  = s_sum[0] * (1.0f / NBLK);
    float var = s_sq[0] * (1.0f / NBLK) - mu * mu;
    float rs  = rsqrtf(var + LN_EPS);
    for (int n = threadIdx.x; n < NBLK; n += 256) {
        float v = (grow[n] - mu) * rs * ln_g[n] + ln_b[n];
        h[(size_t)b * NBLK + n] = (__bf16)v;
    }
}

// ---------------------------------------------------------------------------
// 3) WMMA GEMM:  C(MxN) = A(MxK, bf16 row-major) @ Bt(NxK, bf16 row-major)^T
//    + bias, optional sigmoid. One 16x16 tile per wave; K-loop of
//    v_wmma_f32_16x16x32_bf16 with f32 accumulation.
//
//    A fragment (16-bit A 16x32, ISA 7.12.2): lane m=lane&15,
//      lanes 0-15 hold K {0..7, 16..23}; lanes 16-31 hold K {8..15, 24..31}
//      -> two contiguous v8bf (b128) loads at +half*8 and +half*8+16.
//    B fragment (16-bit B 32x16): lane n=lane&15,
//      lanes 0-15 hold K {0..7, 16..23}; lanes 16-31 hold K {8..15, 24..31}
//      -> two contiguous v8bf loads at +half*8 and +half*8+16.
//    C/D: VGPR r -> row m0 + r + 8*(lane>>4), col n0 + (lane&15).
// ---------------------------------------------------------------------------
__global__ void wmma_gemm_kernel(const __bf16* __restrict__ A,
                                 const __bf16* __restrict__ Bt,
                                 const float*  __restrict__ bias,
                                 __bf16*       __restrict__ outBf,
                                 float*        __restrict__ outF,
                                 int M, int N, int K, int do_sigmoid) {
    int lane = threadIdx.x & 31;
    int wave = threadIdx.x >> 5;
    int n0   = (blockIdx.x * 4 + wave) * 16;
    int m0   = blockIdx.y * 16;
    if (n0 >= N || m0 >= M) return;          // uniform across the wave

    int half = lane >> 4;                    // 0 or 1
    int r15  = lane & 15;
    const __bf16* arow = A  + (size_t)(m0 + r15) * K + half * 8;
    const __bf16* brow = Bt + (size_t)(n0 + r15) * K + half * 8;

    v8f acc = {};
    for (int k = 0; k < K; k += 32) {
        AFrag a, bm;
        a.h[0]  = *(const v8bf*)(arow + k);
        a.h[1]  = *(const v8bf*)(arow + k + 16);
        bm.h[0] = *(const v8bf*)(brow + k);
        bm.h[1] = *(const v8bf*)(brow + k + 16);
        acc = __builtin_amdgcn_wmma_f32_16x16x32_bf16(
            /*neg_a=*/false, a.v, /*neg_b=*/false, bm.v,
            /*c_mod=*/(short)0, acc, /*reuse_a=*/false, /*reuse_b=*/false);
    }

    int   n  = n0 + r15;
    float bv = bias[n];
#pragma unroll
    for (int r = 0; r < 8; ++r) {
        int   m = m0 + r + half * 8;
        float v = acc[r] + bv;
        if (do_sigmoid) v = 1.0f / (1.0f + __expf(-v));
        if (outBf) outBf[(size_t)m * N + n] = (__bf16)v;
        if (outF)  outF[(size_t)m * N + n]  = v;
    }
}

// ---------------------------------------------------------------------------
// 4) Final layer: out[b] = dot(s[b, 0:512], w3) + b3   (OUT = 1)
// ---------------------------------------------------------------------------
__global__ void final_dot_kernel(const float* __restrict__ s,
                                 const float* __restrict__ w3,
                                 const float* __restrict__ b3,
                                 float*       __restrict__ out) {
    __shared__ float red[256];
    int b = blockIdx.x;
    const float* srow = s + (size_t)b * D1;
    float acc = 0.0f;
    for (int j = threadIdx.x; j < D1; j += 256) acc += srow[j] * w3[j];
    red[threadIdx.x] = acc;
    __syncthreads();
    for (int st = 128; st > 0; st >>= 1) {
        if (threadIdx.x < st) red[threadIdx.x] += red[threadIdx.x + st];
        __syncthreads();
    }
    if (threadIdx.x == 0) out[b] = red[0] + b3[0];
}

// ---------------------------------------------------------------------------
// Host-side launcher
// ---------------------------------------------------------------------------
extern "C" void kernel_launch(void* const* d_in, const int* in_sizes, int n_in,
                              void* d_out, int out_size, void* d_ws, size_t ws_size,
                              hipStream_t stream) {
    (void)in_sizes; (void)n_in; (void)out_size; (void)ws_size;

    const float* x    = (const float*)d_in[0];   // (512, 262144)
    const int*   idx  = (const int*)  d_in[1];   // (4096, 64)
    const float* wsp  = (const float*)d_in[2];   // (4096, 64)
    const float* ln_g = (const float*)d_in[3];   // (4096,)
    const float* ln_b = (const float*)d_in[4];   // (4096,)
    const float* w1   = (const float*)d_in[5];   // (4096, 1024)
    const float* b1   = (const float*)d_in[6];   // (1024,)
    const float* w2   = (const float*)d_in[7];   // (1024, 512)
    const float* b2   = (const float*)d_in[8];   // (512,)
    const float* w3   = (const float*)d_in[9];   // (512, 1)
    const float* b3   = (const float*)d_in[10];  // (1,)
    float* out = (float*)d_out;                  // (512, 1)

    // Workspace layout (bytes), MiB-aligned offsets, ~23 MiB total.
    char* ws = (char*)d_ws;
    const size_t MiB = 1ull << 20;
    float*  g     = (float*) (ws + 0 * MiB);   // 512*4096 f32    = 8 MiB
    __bf16* h     = (__bf16*)(ws + 8 * MiB);   // 512*4096 bf16   = 4 MiB
    __bf16* w1t   = (__bf16*)(ws + 12 * MiB);  // 1024*4096 bf16  = 8 MiB
    __bf16* h1    = (__bf16*)(ws + 20 * MiB);  // 512*1024 bf16   = 1 MiB
    __bf16* w2t   = (__bf16*)(ws + 21 * MiB);  // 512*1024 bf16   = 1 MiB
    float*  sact  = (float*) (ws + 22 * MiB);  // 512*512 f32     = 1 MiB

    // 0) Weight conversion/transposition to bf16 NxK ("Bt") layout.
    transpose_to_bf16_kernel<<<(NBLK * D0 + 255) / 256, 256, 0, stream>>>(w1, w1t, NBLK, D0);
    transpose_to_bf16_kernel<<<(D0 * D1 + 255) / 256, 256, 0, stream>>>(w2, w2t, D0, D1);

    // 1) Sparse gather-dot (streams the full 512 MB of x; float4 gathers).
    gather_dot_kernel<<<(BATCH * NBLK) / 256, 256, 0, stream>>>(x, idx, wsp, g);

    // 2) LayerNorm -> bf16 activations.
    layernorm_kernel<<<BATCH, 256, 0, stream>>>(g, ln_g, ln_b, h);

    // 3) h(512x4096) @ w1 -> h1(512x1024) bf16, +b1.
    {
        dim3 grid(D0 / 64, BATCH / 16);      // 4 waves/block, 16 cols each
        wmma_gemm_kernel<<<grid, 128, 0, stream>>>(h, w1t, b1, h1, nullptr,
                                                   BATCH, D0, NBLK, 0);
    }

    // 4) sigmoid(h1(512x1024) @ w2 + b2) -> sact(512x512) f32.
    {
        dim3 grid(D1 / 64, BATCH / 16);
        wmma_gemm_kernel<<<grid, 128, 0, stream>>>(h1, w2t, b2, nullptr, sact,
                                                   BATCH, D1, D0, 1);
    }

    // 5) Final 512-dot per batch row -> out (512 f32).
    final_dot_kernel<<<BATCH, 256, 0, stream>>>(sact, w3, b3, out);
}
